// GPT_21638045237300
// MI455X (gfx1250) — matrix operand order repaired and checked by reading, not compiled, not run on hardware
//
#include <hip/hip_runtime.h>
#include <math.h>

#define DEVINL __device__ __forceinline__

typedef __attribute__((ext_vector_type(16))) __bf16 bf16x16;
typedef __attribute__((ext_vector_type(8)))  float  f32x8;

union Frag16 {
  bf16x16 v;
  unsigned short u[16];
  uint4 q[2];
};

constexpr int Bb  = 2;
constexpr int Tt  = 1024;
constexpr int Ee  = 1024;
constexpr int Hh  = 16;
constexpr int HDd = 64;
constexpr int Ll  = 8;
constexpr int Vv  = 32000;
constexpr int FFf = 4096;
constexpr int Mm  = Bb * Tt;   // 2048 rows

#define NEG_INF (-__builtin_inff())

// ---------------------------------------------------------------------------
// Async global->LDS copy (CDNA5). Prototype (from hipcc diagnostics):
//   (int4 addrspace(1)*, int4 addrspace(3)*, imm offset, imm cpol)
// Guarded: host pass / older toolchains fall back to synchronous uint4 copy.
// ---------------------------------------------------------------------------
#if __has_builtin(__builtin_amdgcn_global_load_async_to_lds_b128) && \
    __has_builtin(__builtin_amdgcn_s_wait_asynccnt)
#define USE_ASYNC 1
#else
#define USE_ASYNC 0
#endif

typedef int i32x4 __attribute__((vector_size(16)));
typedef __attribute__((address_space(1))) i32x4* g4p;
typedef __attribute__((address_space(3))) i32x4* l4p;

DEVINL void cp16(const void* g, void* l) {
#if USE_ASYNC
  // 16B per lane, wave moves 512B per instruction; tracked by ASYNCcnt.
  __builtin_amdgcn_global_load_async_to_lds_b128(
      (g4p)(unsigned long long)g,
      (l4p)(unsigned int)(unsigned long long)l, 0, 0);
#else
  *(uint4*)l = *(const uint4*)g;
#endif
}

template <int N>
DEVINL void wait_async() {
#if USE_ASYNC
  __builtin_amdgcn_s_wait_asynccnt(N);
#endif
}

// fp32 -> bf16 round-to-nearest-even
DEVINL unsigned short f2bf(float f) {
  unsigned int u = __float_as_uint(f);
  u += 0x7FFFu + ((u >> 16) & 1u);
  return (unsigned short)(u >> 16);
}

DEVINL f32x8 zero8() {
  f32x8 z = {0.f, 0.f, 0.f, 0.f, 0.f, 0.f, 0.f, 0.f};
  return z;
}

DEVINL f32x8 wmma_bf16(bf16x16 a, bf16x16 b, f32x8 c) {
  return __builtin_amdgcn_wmma_f32_16x16x32_bf16(false, a, false, b,
                                                 (short)0, c, false, false);
}

// ---------------------------------------------------------------------------
// x[b,t,:] = tok_emb[idx[b,t],:] + pos_emb[t,:]   (fp32, residual stream)
// ---------------------------------------------------------------------------
__global__ __launch_bounds__(256) void k_embed(const int* __restrict__ idx,
                                               const float* __restrict__ tok,
                                               const float* __restrict__ pos,
                                               float* __restrict__ x) {
  const int bt = blockIdx.x;
  const int t  = bt % Tt;
  const int token = idx[bt];
  const float* tp = tok + (size_t)token * Ee;
  const float* pp = pos + (size_t)t * Ee;
  float* xp = x + (size_t)bt * Ee;
  for (int i = threadIdx.x; i < Ee; i += 256) xp[i] = tp[i] + pp[i];
}

// ---------------------------------------------------------------------------
// fp32 -> bf16 bulk convert (per-layer weights; re-read hits 192MB L2)
// ---------------------------------------------------------------------------
__global__ __launch_bounds__(256) void k_cvt(const float* __restrict__ s,
                                             unsigned short* __restrict__ d,
                                             int nquads) {
  const int stride = gridDim.x * 256;
  for (int i = blockIdx.x * 256 + threadIdx.x; i < nquads; i += stride) {
    float4 f = ((const float4*)s)[i];
    uint2 o;
    o.x = (unsigned)f2bf(f.x) | ((unsigned)f2bf(f.y) << 16);
    o.y = (unsigned)f2bf(f.z) | ((unsigned)f2bf(f.w) << 16);
    ((uint2*)d)[i] = o;
  }
}

// ---------------------------------------------------------------------------
// Faithful quirk LN: normalize over axis=1 (the T axis), unbiased (ddof=1).
// Reads fp32 residual stream, writes bf16 (consumed only by WMMA GEMMs).
// Block = 32 e-columns x 8 t-slices. Grid = (E/32, B).
// ---------------------------------------------------------------------------
__global__ __launch_bounds__(256) void k_ln(const float* __restrict__ x,
                                            const float* __restrict__ g,
                                            const float* __restrict__ b,
                                            unsigned short* __restrict__ out) {
  __shared__ float sS[8][32];
  __shared__ float sQ[8][32];
  __shared__ float sMean[32];
  __shared__ float sInv[32];
  const int ec = threadIdx.x & 31;
  const int ts = threadIdx.x >> 5;
  const int bb = blockIdx.y;
  const int e  = blockIdx.x * 32 + ec;
  const float* xp = x + (size_t)bb * Tt * Ee + e;

  float s = 0.f, q = 0.f;
  const int t0 = ts * (Tt / 8);
  for (int t = t0; t < t0 + Tt / 8; ++t) {
    float v = xp[(size_t)t * Ee];
    s += v; q += v * v;
  }
  sS[ts][ec] = s; sQ[ts][ec] = q;
  __syncthreads();
  if (ts == 0) {
    float S = 0.f, Q = 0.f;
    #pragma unroll
    for (int i = 0; i < 8; i++) { S += sS[i][ec]; Q += sQ[i][ec]; }
    float mean = S / (float)Tt;
    float var  = (Q - S * S / (float)Tt) / (float)(Tt - 1);   // ddof=1
    sMean[ec] = mean;
    sInv[ec]  = rsqrtf(var + 1e-5f);
  }
  __syncthreads();
  const float mean = sMean[ec], inv = sInv[ec];
  const float gg = g[e], bv = b[e];
  unsigned short* op = out + (size_t)bb * Tt * Ee + e;
  for (int t = t0; t < t0 + Tt / 8; ++t) {
    op[(size_t)t * Ee] = f2bf(gg * (xp[(size_t)t * Ee] - mean) * inv + bv);
  }
}

// ---------------------------------------------------------------------------
// bf16-WMMA GEMM, double-buffered LDS with async global->LDS staging.
//   A: bf16 [M][K]
//   WMODE 0: W row-major [K][N];  WMODE 1: W in (H, K, 64) QKV layout
//   WBF16 1: W already bf16 (pure byte-copy staging, async)
//   WBF16 0: W fp32 (synchronous stage + convert; used for LM head)
//   OMODE 0: fp32 [M][N];  OMODE 1: bf16 (B,H,T,64);  OMODE 2: bf16 [M][N]
// Block 256 thr (8 waves), tile 128x128xK, k-step 32; wave = 32x64 output.
// ---------------------------------------------------------------------------
template <int WMODE, int OMODE, bool RELU, bool RESID, bool WBF16>
__global__ __launch_bounds__(256) void k_gemm(const unsigned short* __restrict__ A,
                                              const void* __restrict__ Wp,
                                              const float* __restrict__ bias,
                                              const float* __restrict__ resid,
                                              void* __restrict__ Cout,
                                              int N, int K) {
  __shared__ unsigned short As[2][128 * 32];
  __shared__ unsigned short Bs[2][32 * 128];

  const unsigned short* Wb = (const unsigned short*)Wp;
  const float*          Wf = (const float*)Wp;

  const int tid  = threadIdx.x;
  const int lane = tid & 31;
  const int wid  = tid >> 5;
  const int wm   = wid & 3;
  const int wn   = wid >> 2;
  const int rowBase = blockIdx.y * 128;
  const int colBase = blockIdx.x * 128;

  f32x8 acc[2][4];
  #pragma unroll
  for (int i = 0; i < 2; i++)
    #pragma unroll
    for (int j = 0; j < 4; j++) acc[i][j] = zero8();

  // Issue one 128x32 A tile + 32x128 B tile into buffer `buf`.
  auto issue = [&](int k0, int buf) {
    // A: 4096 bf16 = 512 x 16B segs; 2 per thread
    #pragma unroll
    for (int i = 0; i < 2; i++) {
      int slot = tid + i * 256;
      int r = slot >> 2, seg = (slot & 3) << 3;      // 4 segs of 8 elems/row
      cp16(A + (size_t)(rowBase + r) * K + k0 + seg, &As[buf][r * 32 + seg]);
    }
    if (WBF16) {
      #pragma unroll
      for (int i = 0; i < 2; i++) {
        int slot = tid + i * 256;
        int r = slot >> 4, seg = (slot & 15) << 3;   // 16 segs of 8 elems/row
        const unsigned short* src;
        if (WMODE == 0) {
          src = Wb + (size_t)(k0 + r) * N + colBase + seg;
        } else {
          int n = colBase + seg;                     // head-blocked (h, c, d)
          src = Wb + ((size_t)(n >> 6) * K + (k0 + r)) * 64 + (n & 63);
        }
        cp16(src, &Bs[buf][r * 128 + seg]);
      }
    } else {
      // fp32 weights: synchronous stage + convert (LM head path)
      #pragma unroll
      for (int i = 0; i < 4; i++) {
        int s = tid + i * 256;
        int r = s >> 5, cv = (s & 31) << 2;
        float4 f;
        if (WMODE == 0) {
          f = *(const float4*)(Wf + (size_t)(k0 + r) * N + colBase + cv);
        } else {
          int n = colBase + cv;
          f = *(const float4*)(Wf + ((size_t)(n >> 6) * K + (k0 + r)) * 64 + (n & 63));
        }
        unsigned short* d = &Bs[buf][r * 128 + cv];
        d[0] = f2bf(f.x); d[1] = f2bf(f.y); d[2] = f2bf(f.z); d[3] = f2bf(f.w);
      }
    }
  };

  constexpr int ASYNC_PER_TILE = WBF16 ? 4 : 2;  // async insts per wave per tile
  const int nk = K / 32;

  issue(0, 0);
  for (int kt = 0; kt < nk; kt++) {
    if (kt + 1 < nk) {
      issue((kt + 1) * 32, (kt + 1) & 1);
      wait_async<ASYNC_PER_TILE>();          // tile kt complete, kt+1 in flight
    } else {
      wait_async<0>();
    }
    __syncthreads();                         // tile kt visible to all waves

    const unsigned short* Ab = As[kt & 1];
    const unsigned short* Bt = Bs[kt & 1];

    // A fragments (16x32): lane m = lane&15, kb = 0/8 by half-wave
    bf16x16 af[2];
    {
      const int m = lane & 15, kb = (lane < 16) ? 0 : 8;
      #pragma unroll
      for (int mt = 0; mt < 2; mt++) {
        const unsigned short* p = &Ab[(wm * 32 + mt * 16 + m) * 32];
        Frag16 fr;
        fr.q[0] = *(const uint4*)(p + kb);
        fr.q[1] = *(const uint4*)(p + 16 + kb);
        af[mt] = fr.v;
      }
    }
    // B fragments (32x16): N = lane&15; lanes 0-15 K=0..7,16..23, 16-31 K=8..15,24..31
    #pragma unroll
    for (int nt = 0; nt < 4; nt++) {
      const int col = wn * 64 + nt * 16 + (lane & 15);
      const int kb  = (lane < 16) ? 0 : 8;
      Frag16 fr;
      #pragma unroll
      for (int j = 0; j < 16; j++) fr.u[j] = Bt[(kb + (j & 7) + ((j >> 3) << 4)) * 128 + col];
      #pragma unroll
      for (int mt = 0; mt < 2; mt++) acc[mt][nt] = wmma_bf16(af[mt], fr.v, acc[mt][nt]);
    }
    __syncthreads();                         // done reading buf[kt&1]
  }

  // Epilogue: C layout: VGPR r -> M = r + (lane>=16 ? 8 : 0), N = lane&15
  const int hi = lane >> 4, nl = lane & 15;
  #pragma unroll
  for (int mt = 0; mt < 2; mt++) {
    #pragma unroll
    for (int nt = 0; nt < 4; nt++) {
      #pragma unroll
      for (int r = 0; r < 8; r++) {
        const int row = rowBase + wm * 32 + mt * 16 + r + hi * 8;
        const int col = colBase + wn * 64 + nt * 16 + nl;
        float vout = acc[mt][nt][r];
        if (bias) vout += bias[col];
        if (RESID) vout += resid[(size_t)row * N + col];
        if (RELU) vout = fmaxf(vout, 0.f);
        if (OMODE == 0) {
          ((float*)Cout)[(size_t)row * N + col] = vout;
        } else if (OMODE == 1) {
          const int bq = row / Tt, tq = row % Tt;
          ((unsigned short*)Cout)[(((size_t)bq * Hh + (col >> 6)) * Tt + tq) * 64 + (col & 63)] =
              f2bf(vout);
        } else {
          ((unsigned short*)Cout)[(size_t)row * N + col] = f2bf(vout);
        }
      }
    }
  }
}

// ---------------------------------------------------------------------------
// Flash attention (faithful: NO 1/sqrt(hd) scale), causal, online softmax.
// Q/K/V bf16 in (B,H,T,64); O written bf16 directly into (B,T,E) layout.
// Block 128 thr (4 waves); wave = 16 query rows; K/V 64-key blocks staged
// into LDS with async byte-copies.
// ---------------------------------------------------------------------------
__global__ __launch_bounds__(128) void k_attn(const unsigned short* __restrict__ Q,
                                              const unsigned short* __restrict__ K,
                                              const unsigned short* __restrict__ Vp,
                                              unsigned short* __restrict__ O) {
  __shared__ unsigned short Ks[64 * 64];
  __shared__ unsigned short Vs[64 * 64];
  __shared__ unsigned short Ps[4 * 16 * 64];

  const int bh = blockIdx.y;
  const int bb = bh / Hh, hh = bh % Hh;
  const unsigned short* Qb = Q  + (size_t)bh * Tt * 64;
  const unsigned short* Kb = K  + (size_t)bh * Tt * 64;
  const unsigned short* Vb = Vp + (size_t)bh * Tt * 64;

  const int lane = threadIdx.x & 31;
  const int w    = threadIdx.x >> 5;
  const int hi   = lane >> 4, nl = lane & 15;
  const int r0   = blockIdx.x * 64;
  const int rowW = r0 + w * 16;

  // Q as A fragments (16 rows x 64 d): two contiguous 16B chunks per lane
  bf16x16 aq[2];
  {
    const int kb = (lane < 16) ? 0 : 8;
    const unsigned short* p = Qb + (size_t)(rowW + nl) * 64;
    #pragma unroll
    for (int kc = 0; kc < 2; kc++) {
      Frag16 fr;
      fr.q[0] = *(const uint4*)(p + kc * 32 + kb);
      fr.q[1] = *(const uint4*)(p + kc * 32 + 16 + kb);
      aq[kc] = fr.v;
    }
  }

  f32x8 oacc[4];
  #pragma unroll
  for (int nt = 0; nt < 4; nt++) oacc[nt] = zero8();
  float mrow[8], lrow[8];
  #pragma unroll
  for (int r = 0; r < 8; r++) { mrow[r] = NEG_INF; lrow[r] = 0.f; }

  const int nj = blockIdx.x + 1;
  for (int j = 0; j < nj; j++) {
    const int jb = j * 64;
    __syncthreads();                    // previous reads of Ks/Vs complete
    // Stage K_j, V_j: 8KB each = 512 x 16B segs, 4 per thread per buffer
    #pragma unroll
    for (int i = 0; i < 4; i++) {
      int slot = threadIdx.x + i * 128;
      int r = slot >> 3, seg = (slot & 7) << 3;
      cp16(Kb + (size_t)(jb + r) * 64 + seg, &Ks[r * 64 + seg]);
      cp16(Vb + (size_t)(jb + r) * 64 + seg, &Vs[r * 64 + seg]);
    }
    wait_async<0>();
    __syncthreads();

    // S = Q K^T : 16 x 64, contraction over d
    f32x8 st[4];
    #pragma unroll
    for (int nt = 0; nt < 4; nt++) {
      f32x8 s = zero8();
      const int tok = nt * 16 + nl;
      #pragma unroll
      for (int kc = 0; kc < 2; kc++) {
        const int db = kc * 32 + (hi ? 8 : 0);
        Frag16 fr;
        #pragma unroll
        for (int jj = 0; jj < 16; jj++) fr.u[jj] = Ks[tok * 64 + db + (jj & 7) + ((jj >> 3) << 4)];
        s = wmma_bf16(aq[kc], fr.v, s);
      }
      st[nt] = s;
    }

    // Causal mask + online softmax (row stats across 16-lane halves)
    #pragma unroll
    for (int r = 0; r < 8; r++) {
      const int row = rowW + r + hi * 8;
      float mx = NEG_INF;
      #pragma unroll
      for (int nt = 0; nt < 4; nt++) {
        const int key = jb + nt * 16 + nl;
        float sv = st[nt][r];
        if (key > row) sv = NEG_INF;
        st[nt][r] = sv;
        mx = fmaxf(mx, sv);
      }
      #pragma unroll
      for (int off = 1; off < 16; off <<= 1) mx = fmaxf(mx, __shfl_xor(mx, off, 32));
      const float mo = mrow[r];
      const float mn = fmaxf(mo, mx);
      const float sc = (mn == NEG_INF) ? 1.f : __expf(mo - mn);
      float ps = 0.f;
      #pragma unroll
      for (int nt = 0; nt < 4; nt++) {
        const float pv = (mn == NEG_INF) ? 0.f : __expf(st[nt][r] - mn);
        ps += pv;
        Ps[w * 1024 + (r + hi * 8) * 64 + nt * 16 + nl] = f2bf(pv);
      }
      #pragma unroll
      for (int off = 1; off < 16; off <<= 1) ps += __shfl_xor(ps, off, 32);
      lrow[r] = lrow[r] * sc + ps;
      mrow[r] = mn;
      #pragma unroll
      for (int nt = 0; nt < 4; nt++) oacc[nt][r] *= sc;
    }

    // P: C layout -> A fragments via per-wave LDS region (wave-private,
    // DS ops in-order within a wave)
    bf16x16 ap[2];
    {
      const int kb = (lane < 16) ? 0 : 8;
      const unsigned short* p = &Ps[w * 1024 + nl * 64];
      #pragma unroll
      for (int kc = 0; kc < 2; kc++) {
        Frag16 fr;
        fr.q[0] = *(const uint4*)(p + kc * 32 + kb);
        fr.q[1] = *(const uint4*)(p + kc * 32 + 16 + kb);
        ap[kc] = fr.v;
      }
    }
    // O += P (16x64) x V_j (64x64)
    #pragma unroll
    for (int nt = 0; nt < 4; nt++) {
      const int d = nt * 16 + nl;
      #pragma unroll
      for (int kc = 0; kc < 2; kc++) {
        const int kb = kc * 32 + (hi ? 8 : 0);
        Frag16 fr;
        #pragma unroll
        for (int jj = 0; jj < 16; jj++) fr.u[jj] = Vs[(kb + (jj & 7) + ((jj >> 3) << 4)) * 64 + d];
        oacc[nt] = wmma_bf16(ap[kc], fr.v, oacc[nt]);
      }
    }
  }

  // Normalize and write O (bf16) into (B,T,E): col = h*64 + d
  #pragma unroll
  for (int nt = 0; nt < 4; nt++) {
    #pragma unroll
    for (int r = 0; r < 8; r++) {
      const int row = rowW + r + hi * 8;
      const float den = lrow[r];
      const float vo = (den > 0.f) ? oacc[nt][r] / den : 0.f;
      O[((size_t)bb * Tt + row) * Ee + hh * 64 + nt * 16 + nl] = f2bf(vo);
    }
  }
}

// ---------------------------------------------------------------------------
extern "C" void kernel_launch(void* const* d_in, const int* in_sizes, int n_in,
                              void* d_out, int out_size, void* d_ws, size_t ws_size,
                              hipStream_t stream) {
  (void)in_sizes; (void)n_in; (void)out_size; (void)ws_size;
  const int*   idx     = (const int*)  d_in[0];
  const float* tok_emb = (const float*)d_in[1];
  const float* pos_emb = (const float*)d_in[2];
  const float* Wq      = (const float*)d_in[3];
  const float* Wk      = (const float*)d_in[4];
  const float* Wvp     = (const float*)d_in[5];
  const float* projW   = (const float*)d_in[6];
  const float* projB   = (const float*)d_in[7];
  const float* ln1g    = (const float*)d_in[8];
  const float* ln1b    = (const float*)d_in[9];
  const float* ln2g    = (const float*)d_in[10];
  const float* ln2b    = (const float*)d_in[11];
  const float* W1      = (const float*)d_in[12];
  const float* b1      = (const float*)d_in[13];
  const float* W2      = (const float*)d_in[14];
  const float* b2      = (const float*)d_in[15];
  const float* lnfg    = (const float*)d_in[16];
  const float* lnfb    = (const float*)d_in[17];
  const float* headW   = (const float*)d_in[18];
  const float* headB   = (const float*)d_in[19];
  float* out = (float*)d_out;

  // Workspace layout (~52 MB): fp32 residual stream + bf16 activations +
  // one reusable bf16 weight buffer (per-layer conversion, L2-resident).
  char* p = (char*)d_ws;
  float*          x    = (float*)p;          p += (size_t)Mm * Ee * 4;
  unsigned short* hbuf = (unsigned short*)p; p += (size_t)Mm * Ee * 2;
  unsigned short* q    = (unsigned short*)p; p += (size_t)Mm * Ee * 2;
  unsigned short* k    = (unsigned short*)p; p += (size_t)Mm * Ee * 2;
  unsigned short* v    = (unsigned short*)p; p += (size_t)Mm * Ee * 2;
  unsigned short* ao   = (unsigned short*)p; p += (size_t)Mm * Ee * 2;
  unsigned short* ff   = (unsigned short*)p; p += (size_t)Mm * FFf * 2;
  unsigned short* wbuf = (unsigned short*)p; p += (size_t)Ee * FFf * 2;

  const dim3 gE(Ee / 128, Mm / 128);
  const dim3 gF(FFf / 128, Mm / 128);
  const dim3 gV(Vv / 128, Mm / 128);
  const dim3 gLN(Ee / 32, Bb);
  const dim3 gAT(Tt / 64, Bb * Hh);
  const int qEE = Ee * Ee / 4;         // fp32->bf16 quads for an E x E matrix
  const int qEF = Ee * FFf / 4;

  k_embed<<<Bb * Tt, 256, 0, stream>>>(idx, tok_emb, pos_emb, x);

  for (int l = 0; l < Ll; l++) {
    k_ln<<<gLN, 256, 0, stream>>>(x, ln1g + (size_t)l * Ee, ln1b + (size_t)l * Ee, hbuf);

    k_cvt<<<1024, 256, 0, stream>>>(Wq + (size_t)l * Ee * Ee, wbuf, qEE);
    k_gemm<1, 1, false, false, true><<<gE, 256, 0, stream>>>(hbuf, wbuf, nullptr, nullptr, q, Ee, Ee);
    k_cvt<<<1024, 256, 0, stream>>>(Wk + (size_t)l * Ee * Ee, wbuf, qEE);
    k_gemm<1, 1, false, false, true><<<gE, 256, 0, stream>>>(hbuf, wbuf, nullptr, nullptr, k, Ee, Ee);
    k_cvt<<<1024, 256, 0, stream>>>(Wvp + (size_t)l * Ee * Ee, wbuf, qEE);
    k_gemm<1, 1, false, false, true><<<gE, 256, 0, stream>>>(hbuf, wbuf, nullptr, nullptr, v, Ee, Ee);

    k_attn<<<gAT, 128, 0, stream>>>(q, k, v, ao);

    // x = x + ao @ proj_W + proj_b
    k_cvt<<<1024, 256, 0, stream>>>(projW + (size_t)l * Ee * Ee, wbuf, qEE);
    k_gemm<0, 0, false, true, true><<<gE, 256, 0, stream>>>(
        ao, wbuf, projB + (size_t)l * Ee, x, x, Ee, Ee);

    k_ln<<<gLN, 256, 0, stream>>>(x, ln2g + (size_t)l * Ee, ln2b + (size_t)l * Ee, hbuf);

    // ff = relu(h2 @ W1 + b1)  (bf16 out)
    k_cvt<<<1024, 256, 0, stream>>>(W1 + (size_t)l * Ee * FFf, wbuf, qEF);
    k_gemm<0, 2, true, false, true><<<gF, 256, 0, stream>>>(
        hbuf, wbuf, b1 + (size_t)l * FFf, nullptr, ff, FFf, Ee);
    // x = x + ff @ W2 + b2
    k_cvt<<<1024, 256, 0, stream>>>(W2 + (size_t)l * FFf * Ee, wbuf, qEF);
    k_gemm<0, 0, false, true, true><<<gE, 256, 0, stream>>>(
        ff, wbuf, b2 + (size_t)l * Ee, x, x, Ee, FFf);
  }

  k_ln<<<gLN, 256, 0, stream>>>(x, lnfg, lnfb, hbuf);
  // LM head: fp32 weights staged directly (single use; converting 64MB of
  // head weights would add more HBM traffic than it saves)
  k_gemm<0, 0, false, false, false><<<gV, 256, 0, stream>>>(
      hbuf, headW, headB, nullptr, out, Vv, Ee);
}
